// WindowAttention_41343355191387
// MI455X (gfx1250) — matrix operand order repaired and checked by reading, not compiled, not run on hardware
//
#include <hip/hip_runtime.h>

// ---------------------------------------------------------------------------
// Fused Swin window attention for gfx1250 (MI455X), wave32 + WMMA f16.
//   qkv = x @ Wqkv^T + b ; attn = softmax(q k^T * scale + mask) ; o = attn v
//   out = o @ Wproj^T + b
// One window (49x512) per workgroup, 8 waves, all intermediates in LDS.
// GEMM loops ordered (nt, kt, mt[0..3]) so each weight B-fragment is loaded
// once from L2 and reused by 4 WMMAs (4x less L2 traffic than (nt,mt,kt)).
// ---------------------------------------------------------------------------

typedef __attribute__((ext_vector_type(16))) _Float16 v16h;
typedef __attribute__((ext_vector_type(8)))  _Float16 v8h;
typedef __attribute__((ext_vector_type(8)))  float    v8f;

#define DIM        512
#define HEADS      16
#define HEAD_DIM   32
#define NTOK       49
#define NPAD       64
#define NWIN       64
#define SCALE_F    0.17677669529663687f   // 32^-0.5

// LDS layout (bytes):
//   [0,      65536) : X   f16 [64][512]   (x tile; reused as per-wave P scratch)
//   [65536, 131072) : Q   f16 [16][64][32] (pre-scaled q; later overwritten by O)
//   [131072,196608) : K   f16 [16][64][32]
//   [196608,262144) : VT  f16 [16][32][64] (v transposed: [head][d][key])
#define SMEM_BYTES (262144)

// ---- WMMA fragment helpers (contiguous K = 16*(lane/16)+i, same for A and B, wave32) ----------

// A (16x32 f16, M x K), row-major source with row stride `stride` (halfs).
// lane<16: M=lane, K = 0..15; lane>=16: M=lane-16, K = 16..31
__device__ __forceinline__ v16h frag_a(const _Float16* base, int stride,
                                       int m0, int k0, int lane) {
  int m    = m0 + (lane & 15);
  int koff = (lane >> 4) << 4;                 // 0 or 16
  const _Float16* p = base + m * stride + k0 + koff;
  v8h c0 = *(const v8h*)(p);                   // K = koff + 0..7
  v8h c1 = *(const v8h*)(p + 8);               // K = koff + 8..15
  v16h a;
#pragma unroll
  for (int i = 0; i < 8; ++i) { a[i] = c0[i]; a[i + 8] = c1[i]; }
  return a;
}

// B (32x16 f16, K x N) where B[k][n] = src[(n0+n)*stride + k0 + k]
// (src is the N-major operand: row n holds K contiguously).
// lane<16: N=lane, K=0..15 ; lane>=16: N=lane-16, K=16..31  -> 32B contiguous.
__device__ __forceinline__ v16h frag_b(const _Float16* base, int stride,
                                       int n0, int k0, int lane) {
  int n  = n0 + (lane & 15);
  int kb = (lane >> 4) << 4;                   // 0 or 16
  return *(const v16h*)(base + n * stride + k0 + kb);
}

__device__ __forceinline__ v8f wmma_f16(v16h a, v16h b, v8f c) {
  return __builtin_amdgcn_wmma_f32_16x16x32_f16(
      /*neg_a=*/false, a, /*neg_b=*/false, b,
      /*c_mod=*/(short)0, c, /*reuse_a=*/false, /*reuse_b=*/false);
}

// ---- weight f32 -> f16 pre-conversion (tiny, L2-resident afterwards) -------
__global__ void cvt_weights(const float* __restrict__ qkv_w,
                            const float* __restrict__ proj_w,
                            _Float16* __restrict__ wq,
                            _Float16* __restrict__ wp) {
  int i = blockIdx.x * 256 + threadIdx.x;
  if (i < 3 * DIM * DIM) wq[i] = (_Float16)qkv_w[i];
  if (i < DIM * DIM)     wp[i] = (_Float16)proj_w[i];
}

// ---- fused attention kernel ------------------------------------------------
__global__ __launch_bounds__(256)
void win_attn_kernel(const float* __restrict__ x,
                     const float* __restrict__ mask,
                     const _Float16* __restrict__ wqkv,   // [1536][512] f16
                     const float* __restrict__ bqkv,      // [1536]
                     const _Float16* __restrict__ wproj,  // [512][512] f16
                     const float* __restrict__ bproj,     // [512]
                     float* __restrict__ out) {
  const int b    = blockIdx.x;           // window index
  const int tid  = threadIdx.x;
  const int lane = tid & 31;
  const int wave = tid >> 5;             // 0..7

  extern __shared__ __align__(16) char smem[];
  _Float16* X  = (_Float16*)(smem);                 // [64][512]
  _Float16* Q  = (_Float16*)(smem + 65536);         // [16][64][32]
  _Float16* Kp = (_Float16*)(smem + 131072);        // [16][64][32]
  _Float16* VT = (_Float16*)(smem + 196608);        // [16][32][64]

  // ---- Stage 1: x[b] -> f16 LDS tile, rows 49..63 zero ---------------------
  const float2* xb2 = (const float2*)(x + (size_t)b * NTOK * DIM);
  for (int i = tid; i < NPAD * (DIM / 2); i += 256) {
    int row  = i >> 8;                    // 256 float2 per row
    int pidx = i & 255;
    float2 v = (row < NTOK) ? xb2[row * (DIM / 2) + pidx]
                            : make_float2(0.f, 0.f);
    X[row * DIM + pidx * 2]     = (_Float16)v.x;
    X[row * DIM + pidx * 2 + 1] = (_Float16)v.y;
  }
  __syncthreads();

  // ---- Stage 2: QKV GEMM  [64,512] x [512,1536] -> Q/K/VT planes -----------
  for (int nt = wave; nt < 96; nt += 8) {           // 16-wide output tiles
    const int seg  = __builtin_amdgcn_readfirstlane(nt >> 5); // 0=q 1=k 2=v
    const int col0 = nt * 16;
    v8f acc[4] = {};                                // 4 M-tiles, B reused 4x
#pragma unroll 4
    for (int kt = 0; kt < 16; ++kt) {
      v16h bw = frag_b(wqkv, DIM, col0, kt * 32, lane);
#pragma unroll
      for (int mt = 0; mt < 4; ++mt) {
        v16h a = frag_a(X, DIM, mt * 16, kt * 32, lane);
        acc[mt] = wmma_f16(a, bw, acc[mt]);
      }
    }
    const int n  = lane & 15;
    const int mb = (lane >> 4) << 3;
    const int c  = col0 + n;                        // 0..1535
    const float bias = bqkv[c];
    const int h = (c >> 5) & 15, d = c & 31;
    if (seg == 0) {                                 // q: scale, [h][row][d]
      _Float16* dst = Q + h * (NPAD * HEAD_DIM) + d;
#pragma unroll
      for (int mt = 0; mt < 4; ++mt)
#pragma unroll
        for (int r = 0; r < 8; ++r)
          dst[(mt * 16 + mb + r) * HEAD_DIM] =
              (_Float16)((acc[mt][r] + bias) * SCALE_F);
    } else if (seg == 1) {                          // k: [h][row][d]
      _Float16* dst = Kp + h * (NPAD * HEAD_DIM) + d;
#pragma unroll
      for (int mt = 0; mt < 4; ++mt)
#pragma unroll
        for (int r = 0; r < 8; ++r)
          dst[(mt * 16 + mb + r) * HEAD_DIM] = (_Float16)(acc[mt][r] + bias);
    } else {                                        // v transposed: [h][d][key]
      _Float16* dst = VT + h * (HEAD_DIM * NPAD) + d * NPAD;
#pragma unroll
      for (int mt = 0; mt < 4; ++mt)
#pragma unroll
        for (int r = 0; r < 8; ++r)
          dst[mt * 16 + mb + r] = (_Float16)(acc[mt][r] + bias);
    }
  }
  __syncthreads();   // also retires stage-2 reads of X before P reuses it

  // ---- Stage 3: attention, 2 heads per wave --------------------------------
  const float* mwin = mask + (size_t)(b & (NWIN - 1)) * NTOK * NTOK;
  _Float16* P = X + wave * (16 * NPAD);   // per-wave scratch [16][64] f16
#pragma unroll
  for (int hh = 0; hh < 2; ++hh) {
    const int h = wave * 2 + hh;
    _Float16* Qh = Q  + h * (NPAD * HEAD_DIM);
    _Float16* Kh = Kp + h * (NPAD * HEAD_DIM);
    _Float16* Vh = VT + h * (HEAD_DIM * NPAD);
#pragma unroll
    for (int mt = 0; mt < 4; ++mt) {
      // S row-block [16 x 64]: one WMMA per tile (K = D = 32)
      v16h aq = frag_a(Qh, HEAD_DIM, mt * 16, 0, lane);
      v8f s[4];
#pragma unroll
      for (int nt = 0; nt < 4; ++nt) {
        v16h bk = frag_b(Kh, HEAD_DIM, nt * 16, 0, lane);  // B[d][key]=k[key][d]
        v8f z = {};
        s[nt] = wmma_f16(aq, bk, z);
      }
      // mask + softmax; row m lives in one VGPR across a 16-lane half-group
      const int n  = lane & 15;
      const int mb = (lane >> 4) << 3;
#pragma unroll
      for (int r = 0; r < 8; ++r) {
        const int m = mt * 16 + mb + r;
        float vals[4];
        float rmax = -3.0e38f;
#pragma unroll
        for (int nt = 0; nt < 4; ++nt) {
          int col = nt * 16 + n;
          float sv = (m < NTOK && col < NTOK)
                         ? s[nt][r] + mwin[m * NTOK + col]
                         : -1.0e30f;
          vals[nt] = sv;
          rmax = fmaxf(rmax, sv);
        }
#pragma unroll
        for (int off = 1; off < 16; off <<= 1)
          rmax = fmaxf(rmax, __shfl_xor(rmax, off, 16));
        float rsum = 0.f;
#pragma unroll
        for (int nt = 0; nt < 4; ++nt) {
          vals[nt] = __expf(vals[nt] - rmax);
          rsum += vals[nt];
        }
#pragma unroll
        for (int off = 1; off < 16; off <<= 1)
          rsum += __shfl_xor(rsum, off, 16);
        const float inv = 1.0f / rsum;
#pragma unroll
        for (int nt = 0; nt < 4; ++nt)
          P[(mb + r) * NPAD + nt * 16 + n] = (_Float16)(vals[nt] * inv);
      }
      // O[mt rows] = P [16x64] @ V [64x32]; overwrite consumed Q rows with O
#pragma unroll
      for (int dt = 0; dt < 2; ++dt) {
        v8f o = {};
#pragma unroll
        for (int kt = 0; kt < 2; ++kt) {
          v16h ap = frag_a(P, NPAD, 0, kt * 32, lane);
          v16h bv = frag_b(Vh, NPAD, dt * 16, kt * 32, lane); // B[key][d]=VT[d][key]
          o = wmma_f16(ap, bv, o);
        }
#pragma unroll
        for (int r = 0; r < 8; ++r)
          Qh[(mt * 16 + mb + r) * HEAD_DIM + dt * 16 + n] = (_Float16)o[r];
      }
    }
  }
  __syncthreads();

  // ---- Stage 4: out = O [64,512] @ Wproj^T + b, f32 store (rows < 49) ------
  float* ob = out + (size_t)b * NTOK * DIM;
  for (int nt = wave; nt < 32; nt += 8) {
    const int c0 = nt * 16;
    v8f acc[4] = {};                                // 4 M-tiles, B reused 4x
#pragma unroll 4
    for (int kt = 0; kt < 16; ++kt) {
      v16h bw = frag_b(wproj, DIM, c0, kt * 32, lane);
#pragma unroll
      for (int mt = 0; mt < 4; ++mt) {
        // O column block kt*32..kt*32+31 == head kt plane (row-major [64][32])
        v16h a = frag_a(Q + kt * (NPAD * HEAD_DIM), HEAD_DIM, mt * 16, 0, lane);
        acc[mt] = wmma_f16(a, bw, acc[mt]);
      }
    }
    const int n  = lane & 15;
    const int mb = (lane >> 4) << 3;
    const int c  = c0 + n;
    const float bias = bproj[c];
#pragma unroll
    for (int mt = 0; mt < 4; ++mt)
#pragma unroll
      for (int r = 0; r < 8; ++r) {
        int m = mt * 16 + mb + r;
        if (m < NTOK) ob[m * DIM + c] = acc[mt][r] + bias;
      }
  }
}

// ---------------------------------------------------------------------------
extern "C" void kernel_launch(void* const* d_in, const int* in_sizes, int n_in,
                              void* d_out, int out_size, void* d_ws, size_t ws_size,
                              hipStream_t stream) {
  (void)in_sizes; (void)n_in; (void)out_size; (void)ws_size;
  const float* x      = (const float*)d_in[0];
  const float* mask   = (const float*)d_in[1];
  const float* qkv_w  = (const float*)d_in[2];
  const float* qkv_b  = (const float*)d_in[3];
  const float* proj_w = (const float*)d_in[4];
  const float* proj_b = (const float*)d_in[5];
  float* out = (float*)d_out;

  _Float16* wq = (_Float16*)d_ws;                 // 1536*512 f16
  _Float16* wp = wq + 3 * DIM * DIM;              // 512*512  f16

  const int ncvt = 3 * DIM * DIM;                 // 786432
  cvt_weights<<<(ncvt + 255) / 256, 256, 0, stream>>>(qkv_w, proj_w, wq, wp);

  win_attn_kernel<<<2048, 256, SMEM_BYTES, stream>>>(
      x, mask, wq, qkv_b, wp, proj_b, out);
}
